// MultiHeadAttention_52321291599967
// MI455X (gfx1250) — matrix operand order repaired and checked by reading, not compiled, not run on hardware
//
#include <hip/hip_runtime.h>

#define BB 2
#define SS 2048
#define EE 1024
#define HH 16
#define DD 64

typedef __attribute__((ext_vector_type(16))) _Float16 v16h;
typedef __attribute__((ext_vector_type(8)))  float    v8f;
typedef __attribute__((ext_vector_type(4)))  float    v4f;

union Frag16 { v16h v; v4f f[2]; };

// A operand, 16-bit, 16x32 chunk: lane m = lane%16, hf = lane/16.
// K pairs live at halves {8*hf + 0..7} and {16 + 8*hf + 0..7}  -> two b128 loads.
__device__ __forceinline__ v16h load_a_f16(const _Float16* p, int hf) {
  Frag16 u;
  u.f[0] = *(const v4f*)(p + 8 * hf);
  u.f[1] = *(const v4f*)(p + 16 + 8 * hf);
  return u.v;
}
// B operand, 16-bit, 32x16 chunk: lane n = lane%16 holds K = {8*hf + 0..7} and {16 + 8*hf + 0..7}.
__device__ __forceinline__ v16h load_b_f16(const _Float16* p, int hf) {
  Frag16 u;
  u.f[0] = *(const v4f*)(p + 8 * hf);
  u.f[1] = *(const v4f*)(p + 16 + 8 * hf);
  return u.v;
}
// A operand built from f32 source (converted on the fly).
__device__ __forceinline__ v16h load_a_f32(const float* p, int hf) {
  v16h v;
  const float* a = p + 8 * hf;
  const float* b = p + 16 + 8 * hf;
#pragma unroll
  for (int j = 0; j < 8; ++j) { v[j] = (_Float16)a[j]; v[8 + j] = (_Float16)b[j]; }
  return v;
}

__device__ __forceinline__ v8f wmma16(v16h a, v16h b, v8f c) {
  return __builtin_amdgcn_wmma_f32_16x16x32_f16(false, a, false, b, (short)0, c,
                                                false, false);
}

__global__ void cvt_kernel(const float* __restrict__ src, _Float16* __restrict__ dst,
                           int n) {
  int i = blockIdx.x * blockDim.x + threadIdx.x;
  if (i < n) dst[i] = (_Float16)src[i];
}

// Per-head projections: y = x @ W^T, x tile 16x64, W 64x64 f16 row-major.
// Q,K stored [B,H,S,D] f16; V stored transposed [B,H,D,S] f16 (so PV B-operand is contiguous).
__global__ void __launch_bounds__(128)
proj_kernel(const float* __restrict__ query, const float* __restrict__ key,
            const float* __restrict__ value,
            const _Float16* __restrict__ WqH, const _Float16* __restrict__ WkH,
            const _Float16* __restrict__ WvH,
            _Float16* __restrict__ Qh, _Float16* __restrict__ Kh,
            _Float16* __restrict__ Vt) {
  int wave = threadIdx.x >> 5;
  int lane = threadIdx.x & 31;
  int m = lane & 15, hf = lane >> 4;
  int tile = blockIdx.x * 4 + wave;
  const int RT = SS / 16;
  int b = tile / (HH * RT);
  int rem = tile % (HH * RT);
  int h = rem / RT;
  int q0 = (rem % RT) * 16;

  const float* xs[3] = {query, key, value};
  const _Float16* Ws[3] = {WqH, WkH, WvH};
  size_t bh = (size_t)b * HH + h;

#pragma unroll
  for (int mat = 0; mat < 3; ++mat) {
    const float* x = xs[mat] + ((size_t)b * SS + q0 + m) * EE + h * DD;
    v16h a0 = load_a_f32(x, hf);
    v16h a1 = load_a_f32(x + 32, hf);
#pragma unroll
    for (int nt = 0; nt < 4; ++nt) {
      int e = nt * 16 + m;
      const _Float16* wr = Ws[mat] + (size_t)e * DD;  // lane n=e reads W[e][k] contiguous
      v8f c = {};
      c = wmma16(a0, load_b_f16(wr, hf), c);
      c = wmma16(a1, load_b_f16(wr + 32, hf), c);
#pragma unroll
      for (int r = 0; r < 8; ++r) {
        int row = r + 8 * hf;  // C layout: VGPR r -> row r+8*hf, lane%16 -> col
        _Float16 o = (_Float16)c[r];
        if (mat == 0)      Qh[(bh * SS + q0 + row) * DD + e] = o;
        else if (mat == 1) Kh[(bh * SS + q0 + row) * DD + e] = o;
        else               Vt[(bh * DD + e) * SS + q0 + row] = o;
      }
    }
  }
}

// Flash attention: one wave per 16-row Q tile; Sk processed in 32-col chunks.
// Online softmax, f32 running stats/accumulator, P relaid C->A layout via LDS.
__global__ void __launch_bounds__(128)
flash_kernel(const _Float16* __restrict__ Qh, const _Float16* __restrict__ Kh,
             const _Float16* __restrict__ Vt, const unsigned char* __restrict__ mask,
             _Float16* __restrict__ Ctx) {
  __shared__ _Float16 Plds[4][16][32];
  int wave = threadIdx.x >> 5;
  int lane = threadIdx.x & 31;
  int m = lane & 15, hf = lane >> 4;
  int tile = blockIdx.x * 4 + wave;
  const int RT = SS / 16;
  int b = tile / (HH * RT);
  int rem = tile % (HH * RT);
  int h = rem / RT;
  int q0 = (rem % RT) * 16;

  size_t bh = (size_t)b * HH + h;
  const _Float16* qrow = Qh + (bh * SS + q0 + m) * DD;
  v16h aq0 = load_a_f16(qrow, hf);
  v16h aq1 = load_a_f16(qrow + 32, hf);
  const _Float16* kbase = Kh + bh * SS * DD;
  const _Float16* vbase = Vt + bh * DD * SS;

  float mrow[8], lrow[8];
  v8f acc[4] = {};
#pragma unroll
  for (int r = 0; r < 8; ++r) { mrow[r] = -3.0e38f; lrow[r] = 0.f; }

  // Causal loop bound (reference mask is tril); mask bytes still consulted on
  // diagonal-crossing chunks for faithfulness to the mask input.
  for (int kb = 0; kb < q0 + 16; kb += 32) {
    const _Float16* kr0 = kbase + (size_t)(kb + m) * DD;       // B row n -> k = kb+n
    const _Float16* kr1 = kbase + (size_t)(kb + 16 + m) * DD;
    v8f c0 = {}, c1 = {};
    c0 = wmma16(aq0, load_b_f16(kr0, hf), c0);
    c0 = wmma16(aq1, load_b_f16(kr0 + 32, hf), c0);
    c1 = wmma16(aq0, load_b_f16(kr1, hf), c1);
    c1 = wmma16(aq1, load_b_f16(kr1 + 32, hf), c1);

    bool diag = (kb + 31 > q0);
    float p0[8], p1[8];
#pragma unroll
    for (int r = 0; r < 8; ++r) {
      float s0 = c0[r], s1 = c1[r];
      if (diag) {
        size_t qg = (size_t)(q0 + r + 8 * hf);
        if (!mask[qg * SS + kb + m])      s0 = -1.0e20f;  // fill BEFORE scale, as in ref
        if (!mask[qg * SS + kb + 16 + m]) s1 = -1.0e20f;
      }
      s0 *= 0.125f;  // 1/sqrt(64)
      s1 *= 0.125f;
      float mx = fmaxf(s0, s1);  // row reduce across the 16-lane half (xor<16 stays inside)
      mx = fmaxf(mx, __shfl_xor(mx, 1));
      mx = fmaxf(mx, __shfl_xor(mx, 2));
      mx = fmaxf(mx, __shfl_xor(mx, 4));
      mx = fmaxf(mx, __shfl_xor(mx, 8));
      float mn = fmaxf(mrow[r], mx);
      float sc = __expf(mrow[r] - mn);
      float e0 = __expf(s0 - mn);
      float e1 = __expf(s1 - mn);
      float sum = e0 + e1;
      sum += __shfl_xor(sum, 1);
      sum += __shfl_xor(sum, 2);
      sum += __shfl_xor(sum, 4);
      sum += __shfl_xor(sum, 8);
      lrow[r] = lrow[r] * sc + sum;
      mrow[r] = mn;
#pragma unroll
      for (int t = 0; t < 4; ++t) acc[t][r] *= sc;
      p0[r] = e0;
      p1[r] = e1;
    }
    // C layout -> LDS row-major [16][32]
#pragma unroll
    for (int r = 0; r < 8; ++r) {
      Plds[wave][r + 8 * hf][m] = (_Float16)p0[r];
      Plds[wave][r + 8 * hf][16 + m] = (_Float16)p1[r];
    }
    asm volatile("s_wait_dscnt 0" ::: "memory");  // CDNA5 split DS counter wait
    v16h pa = load_a_f16(&Plds[wave][m][0], hf);  // reload in A layout
#pragma unroll
    for (int t = 0; t < 4; ++t) {
      const _Float16* vr = vbase + (size_t)(t * 16 + m) * SS + kb;  // Vt[d][k] contiguous
      acc[t] = wmma16(pa, load_b_f16(vr, hf), acc[t]);
    }
  }

#pragma unroll
  for (int r = 0; r < 8; ++r) {
    float inv = 1.0f / lrow[r];
    size_t row = (size_t)q0 + r + 8 * hf;
#pragma unroll
    for (int t = 0; t < 4; ++t)
      Ctx[((size_t)b * SS + row) * EE + h * DD + t * 16 + m] =
          (_Float16)(acc[t][r] * inv);
  }
}

// out = Ctx @ Wo^T + bo   (4096 x 1024 x 1024), one 16x16 tile per wave, K in 32-chunks.
__global__ void __launch_bounds__(128)
outproj_kernel(const _Float16* __restrict__ Ctx, const _Float16* __restrict__ WoH,
               const float* __restrict__ bo, float* __restrict__ out) {
  int wave = threadIdx.x >> 5;
  int lane = threadIdx.x & 31;
  int m = lane & 15, hf = lane >> 4;
  int tile = blockIdx.x * 4 + wave;
  const int CT = EE / 16;
  int rt = tile / CT;
  int ct = tile % CT;

  const _Float16* arow = Ctx + ((size_t)rt * 16 + m) * EE;
  int e = ct * 16 + m;
  const _Float16* brow = WoH + (size_t)e * EE;  // lane n=e reads Wo[e][k] contiguous
  v8f c = {};
#pragma unroll 4
  for (int kc = 0; kc < EE; kc += 32)
    c = wmma16(load_a_f16(arow + kc, hf), load_b_f16(brow + kc, hf), c);

  float bias = bo[e];
#pragma unroll
  for (int r = 0; r < 8; ++r)
    out[((size_t)rt * 16 + r + 8 * hf) * EE + e] = c[r] + bias;
}

extern "C" void kernel_launch(void* const* d_in, const int* in_sizes, int n_in,
                              void* d_out, int out_size, void* d_ws, size_t ws_size,
                              hipStream_t stream) {
  // setup_inputs order: key, query, value, mask, Wq, Wk, Wv, Wo, bo
  const float* key = (const float*)d_in[0];
  const float* query = (const float*)d_in[1];
  const float* value = (const float*)d_in[2];
  const unsigned char* mask = (const unsigned char*)d_in[3];
  const float* Wq = (const float*)d_in[4];
  const float* Wk = (const float*)d_in[5];
  const float* Wv = (const float*)d_in[6];
  const float* Wo = (const float*)d_in[7];
  const float* bo = (const float*)d_in[8];
  float* out = (float*)d_out;

  // Workspace layout (f16 elements): WqH WkH WvH | WoH | Qh | Kh | Vt | Ctx  (~34 MB)
  _Float16* w = (_Float16*)d_ws;
  const size_t DW = (size_t)DD * DD;        // 4096
  const size_t EW = (size_t)EE * EE;        // 1048576
  const size_t HS = (size_t)BB * HH * SS * DD;  // 4194304
  _Float16* WqH = w;
  _Float16* WkH = WqH + DW;
  _Float16* WvH = WkH + DW;
  _Float16* WoH = WvH + DW;
  _Float16* Qh = WoH + EW;
  _Float16* Kh = Qh + HS;
  _Float16* Vt = Kh + HS;
  _Float16* Ctx = Vt + HS;

  cvt_kernel<<<(int)((DW + 255) / 256), 256, 0, stream>>>(Wq, WqH, (int)DW);
  cvt_kernel<<<(int)((DW + 255) / 256), 256, 0, stream>>>(Wk, WkH, (int)DW);
  cvt_kernel<<<(int)((DW + 255) / 256), 256, 0, stream>>>(Wv, WvH, (int)DW);
  cvt_kernel<<<(int)((EW + 255) / 256), 256, 0, stream>>>(Wo, WoH, (int)EW);

  int tiles = BB * HH * (SS / 16);  // 4096 q-tiles, 4 waves/block
  proj_kernel<<<tiles / 4, 128, 0, stream>>>(query, key, value, WqH, WkH, WvH, Qh, Kh,
                                             Vt);
  flash_kernel<<<tiles / 4, 128, 0, stream>>>(Qh, Kh, Vt, mask, Ctx);
  int otiles = (BB * SS / 16) * (EE / 16);  // 16384 output tiles
  outproj_kernel<<<otiles / 4, 128, 0, stream>>>(Ctx, WoH, bo, out);
}
